// ModelXToResult_13829794693341
// MI455X (gfx1250) — matrix operand order repaired and checked by reading, not compiled, not run on hardware
//
#include <hip/hip_runtime.h>
#include <hip/hip_bf16.h>

typedef __attribute__((ext_vector_type(8)))  _Float16 v8h;
typedef __attribute__((ext_vector_type(16))) _Float16 v16h;
typedef __attribute__((ext_vector_type(8)))  float    v8f;

constexpr int HID      = 256;
constexpr int STRIDE   = 264;              // padded row stride in halfs (16B-aligned rows)
constexpr int TILE_PTS = 16;               // points per block
constexpr int BLOCK    = 128;              // 4 waves (wave32)

// ---------------------------------------------------------------------------
// Prep: convert fp32 weight (K x N, row-major) -> fp16 transposed (N x K)
// so WMMA B-fragments are two contiguous 16B loads.
// ---------------------------------------------------------------------------
__global__ void prep_transpose_f16(const float* __restrict__ src,
                                   _Float16* __restrict__ dst) {
    int idx = blockIdx.x * blockDim.x + threadIdx.x;   // 0 .. 65535
    int k = idx >> 8;
    int n = idx & 255;
    dst[n * HID + k] = (_Float16)src[k * HID + n];
}

// A fragment (16x32 f16): lane holds row m = lane%16.
// VGPR r<4 -> K = k0 + 8*half + {2r,2r+1} ; r>=4 -> +16.
__device__ __forceinline__ v16h load_a_frag(const _Float16* H, int lm, int k0, int lh) {
    const _Float16* p = H + lm * STRIDE + k0 + 8 * lh;
    v8h lo = *(const v8h*)(p);
    v8h hi = *(const v8h*)(p + 16);
    return __builtin_shufflevector(lo, hi, 0,1,2,3,4,5,6,7,8,9,10,11,12,13,14,15);
}

// B fragment (32x16 f16) from N-major fp16 weights: lane covers column n0+lane%16,
// K = k0 + 8*half + {0..7}, then +16 -> two 16B loads (matches A).
__device__ __forceinline__ v16h load_b_frag(const _Float16* WT, int n0, int k0, int lm, int lh) {
    const _Float16* p = WT + (size_t)(n0 + lm) * HID + k0 + 8 * lh;
    v8h lo = *(const v8h*)(p);
    v8h hi = *(const v8h*)(p + 16);
    return __builtin_shufflevector(lo, hi, 0,1,2,3,4,5,6,7,8,9,10,11,12,13,14,15);
}

__device__ __forceinline__ v8f wmma_f16(v16h a, v16h b, v8f c) {
    return __builtin_amdgcn_wmma_f32_16x16x32_f16(false, a, false, b,
                                                  (short)0, c, false, false);
}

__global__ void __launch_bounds__(BLOCK)
pinn_elastic_kernel(const float* __restrict__ X,
                    const float* __restrict__ W1, const float* __restrict__ b1,
                    const float* __restrict__ b2, const float* __restrict__ b3,
                    const float* __restrict__ Wo, const float* __restrict__ bo,
                    const _Float16* __restrict__ W2T,
                    const _Float16* __restrict__ W3T,
                    float* __restrict__ out, long long npts) {
    constexpr float NU  = 0.3f;
    constexpr float C11 = 1.0f / (1.0f - NU * NU);
    constexpr float C12 = NU / (1.0f - NU * NU);
    constexpr float C33 = 1.0f / (2.0f * (1.0f + NU));

    // streams: 0 = primal h, 1 = d/dx tangent, 2 = d/dy tangent
    __shared__ __align__(16) _Float16 sbuf[2][3][TILE_PTS * STRIDE];   // ping-pong
    __shared__ float dots[TILE_PTS][6];

    const int tid  = threadIdx.x;
    const int wave = tid >> 5;
    const int lane = tid & 31;
    const int lm   = lane & 15;
    const int lh   = lane >> 4;
    const long long base_pt = (long long)blockIdx.x * TILE_PTS;

    // ---------------- Layer 1: (x,y) -> 256, VALU (tiny) -------------------
    {
        int m  = tid >> 3;                 // 16 points x 8 threads each
        int j0 = tid & 7;
        float x = X[2 * (base_pt + m) + 0];
        float y = X[2 * (base_pt + m) + 1];
        for (int k = 0; k < 32; ++k) {
            int j = j0 + 8 * k;            // coalesced across the 8 threads
            float w0 = W1[j], w1 = W1[HID + j];
            float z = fmaf(x, w0, fmaf(y, w1, b1[j]));
            float h = tanhf(z);
            float g = 1.0f - h * h;
            sbuf[0][0][m * STRIDE + j] = (_Float16)h;
            sbuf[0][1][m * STRIDE + j] = (_Float16)(g * w0);   // tangent wrt x
            sbuf[0][2][m * STRIDE + j] = (_Float16)(g * w1);   // tangent wrt y
        }
    }
    __syncthreads();

    // ---------------- Layers 2 & 3: 256x256 via WMMA -----------------------
    // Each wave owns N-tiles {wave, wave+4, wave+8, wave+12}; processing the
    // primal and tangent tiles of the same N-tiles in the same wave keeps
    // g = 1-h^2 in registers (C-fragment layouts match) => 1 barrier/layer.
    for (int layer = 0; layer < 2; ++layer) {
        const _Float16* WT   = layer ? W3T : W2T;
        const float*    bias = layer ? b3  : b2;
        const _Float16 (*inb)[TILE_PTS * STRIDE]  = sbuf[layer & 1];
        _Float16       (*outb)[TILE_PTS * STRIDE] = sbuf[(layer & 1) ^ 1];

        for (int pair = 0; pair < 2; ++pair) {
            const int n0A = (wave + 8 * pair) * 16;
            const int n0B = (wave + 8 * pair + 4) * 16;

            // ---- primal pass: shared A, two N-tiles ----
            v8f accA = {}, accB = {};
#pragma unroll
            for (int kb = 0; kb < 8; ++kb) {
                int k0 = kb * 32;
                v16h a  = load_a_frag(inb[0], lm, k0, lh);
                v16h bA = load_b_frag(WT, n0A, k0, lm, lh);
                v16h bB = load_b_frag(WT, n0B, k0, lm, lh);
                accA = wmma_f16(a, bA, accA);
                accB = wmma_f16(a, bB, accB);
            }
            v8f gA, gB;
            {
                float bnA = bias[n0A + lm];
                float bnB = bias[n0B + lm];
#pragma unroll
                for (int c = 0; c < 8; ++c) {
                    int m = c + 8 * lh;
                    float hA = tanhf(accA[c] + bnA);
                    float hB = tanhf(accB[c] + bnB);
                    gA[c] = 1.0f - hA * hA;
                    gB[c] = 1.0f - hB * hB;
                    outb[0][m * STRIDE + n0A + lm] = (_Float16)hA;
                    outb[0][m * STRIDE + n0B + lm] = (_Float16)hB;
                }
            }

            // ---- tangent pass: 2 streams x 2 N-tiles, shared A and B ----
            v8f t1A = {}, t1B = {}, t2A = {}, t2B = {};
#pragma unroll
            for (int kb = 0; kb < 8; ++kb) {
                int k0 = kb * 32;
                v16h a1 = load_a_frag(inb[1], lm, k0, lh);
                v16h a2 = load_a_frag(inb[2], lm, k0, lh);
                v16h bA = load_b_frag(WT, n0A, k0, lm, lh);
                v16h bB = load_b_frag(WT, n0B, k0, lm, lh);
                t1A = wmma_f16(a1, bA, t1A);
                t1B = wmma_f16(a1, bB, t1B);
                t2A = wmma_f16(a2, bA, t2A);
                t2B = wmma_f16(a2, bB, t2B);
            }
#pragma unroll
            for (int c = 0; c < 8; ++c) {
                int m = c + 8 * lh;
                outb[1][m * STRIDE + n0A + lm] = (_Float16)(gA[c] * t1A[c]);
                outb[1][m * STRIDE + n0B + lm] = (_Float16)(gB[c] * t1B[c]);
                outb[2][m * STRIDE + n0A + lm] = (_Float16)(gA[c] * t2A[c]);
                outb[2][m * STRIDE + n0B + lm] = (_Float16)(gB[c] * t2B[c]);
            }
        }
        __syncthreads();
    }

    // ---------------- Output layer 256 -> 2 + stress algebra ---------------
    if (tid < TILE_PTS * 6) {
        int m = tid / 6;
        int d = tid % 6;         // 0:u 1:v 2:u_x 3:v_x 4:u_y 5:v_y
        int s = d >> 1;
        int col = d & 1;
        const _Float16* hrow = sbuf[0][s] + m * STRIDE;   // final activations
        float acc = 0.0f;
#pragma unroll 4
        for (int k = 0; k < HID; ++k)
            acc = fmaf((float)hrow[k], Wo[2 * k + col], acc);
        if (s == 0) acc += bo[col];
        dots[m][d] = acc;
    }
    __syncthreads();

    if (tid < TILE_PTS) {
        int m = tid;
        float u  = dots[m][0], v  = dots[m][1];
        float ux = dots[m][2], vx = dots[m][3];
        float uy = dots[m][4], vy = dots[m][5];
        float uxy = uy + vx;
        float sx  = C11 * ux + C12 * vy;
        float sy  = C12 * ux + C11 * vy;
        float sxy = C33 * uxy;
        float sed = 0.5f * (sx * ux + sy * vy + sxy * uxy);
        size_t p = (size_t)(base_pt + m);
        size_t N = (size_t)npts;
        out[0 * N + p] = u;
        out[1 * N + p] = v;
        out[2 * N + p] = sx;
        out[3 * N + p] = sy;
        out[4 * N + p] = sxy;
        out[5 * N + p] = sed;
    }
}

extern "C" void kernel_launch(void* const* d_in, const int* in_sizes, int n_in,
                              void* d_out, int out_size, void* d_ws, size_t ws_size,
                              hipStream_t stream) {
    const float* X  = (const float*)d_in[0];
    const float* W1 = (const float*)d_in[1];
    const float* b1 = (const float*)d_in[2];
    const float* W2 = (const float*)d_in[3];
    const float* b2 = (const float*)d_in[4];
    const float* W3 = (const float*)d_in[5];
    const float* b3 = (const float*)d_in[6];
    const float* Wo = (const float*)d_in[7];
    const float* bo = (const float*)d_in[8];

    _Float16* W2T = (_Float16*)d_ws;
    _Float16* W3T = W2T + (size_t)HID * HID;

    // fp16 transposed weights (N-major) for contiguous WMMA B-fragments
    prep_transpose_f16<<<HID * HID / 256, 256, 0, stream>>>(W2, W2T);
    prep_transpose_f16<<<HID * HID / 256, 256, 0, stream>>>(W3, W3T);

    long long npts = (long long)in_sizes[0] / 2;     // (B*N) points
    int blocks = (int)(npts / TILE_PTS);
    pinn_elastic_kernel<<<blocks, BLOCK, 0, stream>>>(
        X, W1, b1, b2, b3, Wo, bo, W2T, W3T, (float*)d_out, npts);
}
